// AutoQVLALinearW4A4QuaRot_56581899157844
// MI455X (gfx1250) — 8-bit matrix operand order repaired and checked by reading, compiled, not run on hardware
//
#include <hip/hip_runtime.h>
#include <hip/hip_bf16.h>

typedef __attribute__((ext_vector_type(16))) int   v16i;
typedef __attribute__((ext_vector_type(8)))  float v8f;

#define EPS_F 1e-6f

__device__ __constant__ unsigned char kFp8Lut[16] = {
    0xD0, 0xCE, 0xCC, 0xCA, 0xC8, 0xC4, 0xC0, 0xB8,
    0x00, 0x38, 0x40, 0x44, 0x48, 0x4A, 0x4C, 0x4E
};

__device__ __forceinline__ unsigned char quant_fp8(float v, float inv_scale) {
    float r = rintf(v * inv_scale);
    r = fminf(fmaxf(r, -8.0f), 7.0f);
    return kFp8Lut[(int)r + 8];
}

__global__ __launch_bounds__(256) void wquant_kernel(const float* __restrict__ w,
                                                     unsigned char* __restrict__ qw,
                                                     float* __restrict__ wsc, int K) {
    __shared__ float red[8];
    const int n = blockIdx.x;
    const float* wr = w + (size_t)n * K;
    float amax = 0.0f;
    for (int i = threadIdx.x; i < K; i += 256)
        amax = fmaxf(amax, fabsf(wr[i]));
    for (int o = 16; o > 0; o >>= 1)
        amax = fmaxf(amax, __shfl_xor(amax, o, 32));
    if ((threadIdx.x & 31) == 0) red[threadIdx.x >> 5] = amax;
    __syncthreads();
    if (threadIdx.x == 0) {
        float m = red[0];
        for (int i = 1; i < 8; ++i) m = fmaxf(m, red[i]);
        red[0] = fmaxf(m, EPS_F) * (1.0f / 7.0f);
    }
    __syncthreads();
    const float scale = red[0];
    const float inv = 1.0f / scale;
    unsigned char* qr = qw + (size_t)n * K;
    for (int i = threadIdx.x; i < K; i += 256)
        qr[i] = quant_fp8(wr[i], inv);
    if (threadIdx.x == 0) wsc[n] = scale;
}

__global__ __launch_bounds__(256) void fwht_quant_kernel(const float* __restrict__ x,
                                                         unsigned char* __restrict__ qx,
                                                         float* __restrict__ sx, int K) {
    __shared__ float s[4096];
    __shared__ float red[8];
    const int row = blockIdx.x;
    const float* xr = x + (size_t)row * K;
    for (int i = threadIdx.x; i < K; i += 256) s[i] = xr[i];
    __syncthreads();
    for (int h = 1; h < K; h <<= 1) {
        for (int i = threadIdx.x; i < (K >> 1); i += 256) {
            int ia = ((i & ~(h - 1)) << 1) | (i & (h - 1));
            int ib = ia + h;
            float a = s[ia], b = s[ib];
            s[ia] = a + b;
            s[ib] = a - b;
        }
        __syncthreads();
    }
    float amax = 0.0f;
    for (int i = threadIdx.x; i < K; i += 256) {
        float v = s[i] * (1.0f / 64.0f);
        s[i] = v;
        amax = fmaxf(amax, fabsf(v));
    }
    for (int o = 16; o > 0; o >>= 1)
        amax = fmaxf(amax, __shfl_xor(amax, o, 32));
    if ((threadIdx.x & 31) == 0) red[threadIdx.x >> 5] = amax;
    __syncthreads();
    if (threadIdx.x == 0) {
        float m = red[0];
        for (int i = 1; i < 8; ++i) m = fmaxf(m, red[i]);
        red[0] = fmaxf(m, EPS_F) * (1.0f / 7.0f);
    }
    __syncthreads();
    const float scale = red[0];
    const float inv = 1.0f / scale;
    unsigned char* qr = qx + (size_t)row * K;
    for (int i = threadIdx.x; i < K; i += 256)
        qr[i] = quant_fp8(s[i], inv);
    if (threadIdx.x == 0) sx[row] = scale;
}

#define BROW_STRIDE 144
__global__ __launch_bounds__(256) void w4a4_gemm_kernel(
    const unsigned char* __restrict__ qx, const unsigned char* __restrict__ qw,
    const float* __restrict__ sx, const float* __restrict__ wsc,
    const float* __restrict__ bias, float* __restrict__ y,
    int M, int N, int K) {
    __shared__ unsigned char bbuf[2][64 * BROW_STRIDE];

    const int lane  = threadIdx.x & 31;
    const int wv    = threadIdx.x >> 5;
    const int nblk  = N >> 6;
    const int bm    = blockIdx.x / nblk;
    const int bn    = blockIdx.x % nblk;
    const int m_tile = bm * 256 + wv * 32;
    const int n_base = bn * 64;
    const bool hi   = lane >= 16;
    const int  l    = lane & 15;

    const unsigned char* arow0 = qx + (size_t)(m_tile + l) * K + (hi ? 16 : 0);
    const unsigned char* arow1 = arow0 + (size_t)16 * K;
    const unsigned char* bsrc  = qw + (size_t)n_base * K;
    const unsigned lds0 = (unsigned)(size_t)&bbuf[0][0];

    auto issue_copy = [&](int chunk, int buf) {
#pragma unroll
        for (int s2 = 0; s2 < 2; ++s2) {
            const int seg = threadIdx.x + (s2 << 8);
            const int nr  = seg >> 3;
            const int of  = (seg & 7) << 4;
            const unsigned lds = lds0 + (unsigned)(buf * (64 * BROW_STRIDE) +
                                                   nr * BROW_STRIDE + of);
            const unsigned char* g = bsrc + (size_t)nr * K + (size_t)chunk * 128 + of;
            asm volatile("global_load_async_to_lds_b128 %0, %1, off"
                         :: "v"(lds), "v"(g) : "memory");
        }
    };

    auto load_a = [&](const unsigned char* arow, size_t kc) -> v16i {
        v16i a;
        const int2* ap = (const int2*)(arow + kc);
#pragma unroll
        for (int j = 0; j < 4; ++j) {
            int2 v = ap[(j >> 1) * 4 + (j & 1)];
            a[2 * j]     = v.x;
            a[2 * j + 1] = v.y;
        }
        const int2* ap2 = (const int2*)(arow + kc + 64);
#pragma unroll
        for (int j = 0; j < 4; ++j) {
            int2 v = ap2[(j >> 1) * 4 + (j & 1)];
            a[8 + 2 * j] = v.x;
            a[9 + 2 * j] = v.y;
        }
        return a;
    };

    auto load_b = [&](const unsigned char* bb, int t) -> v16i {
        const unsigned char* bp = bb + (t * 16 + l) * BROW_STRIDE + (hi ? 16 : 0);
        v16i b;
#pragma unroll
        for (int q = 0; q < 4; ++q) {
            int4 v = *(const int4*)(bp + 32 * q);
            b[4 * q]     = v.x;
            b[4 * q + 1] = v.y;
            b[4 * q + 2] = v.z;
            b[4 * q + 3] = v.w;
        }
        return b;
    };

    v8f acc0[4], acc1[4];
#pragma unroll
    for (int t = 0; t < 4; ++t) { acc0[t] = (v8f){}; acc1[t] = (v8f){}; }

    const int nch = K >> 7;
    issue_copy(0, 0);

    for (int i = 0; i < nch - 1; ++i) {
        const int cur = i & 1;
        const size_t kc = (size_t)i * 128;

        __builtin_prefetch(arow0 + kc + 256, 0, 3);
        __builtin_prefetch(arow1 + kc + 256, 0, 3);
        v16i a0 = load_a(arow0, kc);
        v16i a1 = load_a(arow1, kc);

        issue_copy(i + 1, cur ^ 1);
        asm volatile("s_wait_asynccnt 0x2" ::: "memory");
        __syncthreads();

        const unsigned char* bb = &bbuf[cur][0];
#pragma unroll
        for (int t = 0; t < 4; ++t) {
            v16i b = load_b(bb, t);
            acc0[t] = __builtin_amdgcn_wmma_f32_16x16x128_fp8_fp8(
                a0, b, (short)0, acc0[t], false, false);
            acc1[t] = __builtin_amdgcn_wmma_f32_16x16x128_fp8_fp8(
                a1, b, (short)0, acc1[t], false, false);
        }
        __syncthreads();
    }

    {
        const int i = nch - 1;
        const int cur = i & 1;
        const size_t kc = (size_t)i * 128;
        v16i a0 = load_a(arow0, kc);
        v16i a1 = load_a(arow1, kc);
        asm volatile("s_wait_asynccnt 0x0" ::: "memory");
        __syncthreads();
        const unsigned char* bb = &bbuf[cur][0];
#pragma unroll
        for (int t = 0; t < 4; ++t) {
            v16i b = load_b(bb, t);
            acc0[t] = __builtin_amdgcn_wmma_f32_16x16x128_fp8_fp8(
                a0, b, (short)0, acc0[t], false, false);
            acc1[t] = __builtin_amdgcn_wmma_f32_16x16x128_fp8_fp8(
                a1, b, (short)0, acc1[t], false, false);
        }
    }

    float sxv0[8], sxv1[8];
#pragma unroll
    for (int r = 0; r < 8; ++r) {
        sxv0[r] = sx[m_tile + r + (hi ? 8 : 0)];
        sxv1[r] = sx[m_tile + 16 + r + (hi ? 8 : 0)];
    }
#pragma unroll
    for (int t = 0; t < 4; ++t) {
        const int n  = n_base + t * 16 + l;
        const float wn = wsc[n];
        const float bn = bias[n];
#pragma unroll
        for (int r = 0; r < 8; ++r) {
            const int m = m_tile + r + (hi ? 8 : 0);
            y[(size_t)m * N + n]        = acc0[t][r] * sxv0[r] * wn + bn;
            y[(size_t)(m + 16) * N + n] = acc1[t][r] * sxv1[r] * wn + bn;
        }
    }
}

extern "C" void kernel_launch(void* const* d_in, const int* in_sizes, int n_in,
                              void* d_out, int out_size, void* d_ws, size_t ws_size,
                              hipStream_t stream) {
    const float* x    = (const float*)d_in[0];
    const float* w    = (const float*)d_in[1];
    const float* bias = (const float*)d_in[2];
    float* y = (float*)d_out;

    const int N = in_sizes[2];
    const int K = in_sizes[1] / N;
    const int M = in_sizes[0] / K;

    char* ws = (char*)d_ws;
    unsigned char* qx  = (unsigned char*)ws;
    unsigned char* qw  = (unsigned char*)(ws + (size_t)M * K);
    float*         sx  = (float*)(ws + (size_t)M * K + (size_t)N * K);
    float*         wsc = sx + M;

    wquant_kernel<<<N, 256, 0, stream>>>(w, qw, wsc, K);
    fwht_quant_kernel<<<M, 256, 0, stream>>>(x, qx, sx, K);

    const int blocks = (M / 256) * (N / 64);
    w4a4_gemm_kernel<<<blocks, 256, 0, stream>>>(qx, qw, sx, wsc, bias, y, M, N, K);
}
